// TopKRouter_35759897706713
// MI455X (gfx1250) — matrix operand order repaired and checked by reading, not compiled, not run on hardware
//
#include <hip/hip_runtime.h>

// ---------------------------------------------------------------------------
// Top-2 MoE router for MI455X (gfx1250, wave32).
//   logits = hidden[T,2048] (fp32) x weight[64,2048]^T via V_WMMA_F32_16X16X32_BF16
//   (bf16 inputs, fp32 accumulate), fused per-token top-2 + 2-way softmax.
// Entire bf16 weight matrix staged once per workgroup into the 320KB WGP LDS.
// ---------------------------------------------------------------------------

#define H_DIM   2048
#define E_DIM   64
#define TOPK    2
#define LDSS    2056          // padded LDS row stride: 1028 dwords % 64 banks = 4
#define BLOCK   512
#define WAVES   (BLOCK / 32)

typedef __attribute__((ext_vector_type(16))) __bf16 v16bf;
typedef __attribute__((ext_vector_type(2)))  __bf16 v2bf;
typedef __attribute__((ext_vector_type(8)))  float  v8f;
typedef __attribute__((ext_vector_type(4)))  float  f32x4;

union BF16x16 {
    v16bf    v;
    unsigned u[8];
    uint4    q[2];
};

// Pack two fp32 -> two bf16 (RNE) in one dword.
__device__ __forceinline__ unsigned pack_bf16(float a, float b) {
#if __has_builtin(__builtin_amdgcn_cvt_pk_bf16_f32)
    v2bf r = __builtin_amdgcn_cvt_pk_bf16_f32(a, b);   // v_cvt_pk_bf16_f32
    return __builtin_bit_cast(unsigned, r);
#else
    unsigned ua = __builtin_bit_cast(unsigned, a);
    unsigned ub = __builtin_bit_cast(unsigned, b);
    ua += 0x7fffu + ((ua >> 16) & 1u);
    ub += 0x7fffu + ((ub >> 16) & 1u);
    return (ua >> 16) | (ub & 0xffff0000u);
#endif
}

// Sortable key: monotone float map in [63:32], (63-expert) in [31:0] so that
// on equal logits the larger key is the LOWER expert index (top_k tie rule).
__device__ __forceinline__ unsigned long long mk_key(float v, int e) {
    unsigned u = __builtin_bit_cast(unsigned, v);
    u = (u & 0x80000000u) ? ~u : (u | 0x80000000u);
    return ((unsigned long long)u << 32) | (unsigned)(63 - e);
}
__device__ __forceinline__ float key_val(unsigned long long k) {
    unsigned u = (unsigned)(k >> 32);
    u = (u & 0x80000000u) ? (u & 0x7fffffffu) : ~u;
    return __builtin_bit_cast(float, u);
}
__device__ __forceinline__ int key_idx(unsigned long long k) {
    return 63 - (int)(k & 0xffffffffu);
}

__global__ void __launch_bounds__(BLOCK)
topk_router_kernel(const float* __restrict__ hidden,   // [T, 2048] fp32
                   const float* __restrict__ weight,   // [64, 2048] fp32
                   float* __restrict__ probs,          // [T, 2] fp32
                   int*   __restrict__ indices,        // [T, 2] int32
                   int numTokens)
{
    // 64 rows x 2056 elems x 2B = 257 KB of the 320 KB WGP LDS.
    __shared__ unsigned short wlds[E_DIM * LDSS];

    // ---- Stage weight fp32 -> bf16 into LDS (one pass, pair-packed) ----
    for (int p = threadIdx.x; p < (E_DIM * H_DIM) / 2; p += BLOCK) {
        const int e = p >> 10;            // 1024 pairs per row
        const int h = (p & 1023) << 1;
        const float2 f = *(const float2*)(weight + e * H_DIM + h);
        *(unsigned*)(&wlds[e * LDSS + h]) = pack_bf16(f.x, f.y);
    }
    __syncthreads();

    const int lane   = threadIdx.x & 31;
    const int wave   = threadIdx.x >> 5;
    const int lane15 = lane & 15;
    const int hi     = lane >> 4;         // half-wave id (K-split per ISA A/B layouts)
    const int numTiles = numTokens >> 4;  // 16 tokens per wave tile

    for (int tile = blockIdx.x * WAVES + wave; tile < numTiles;
         tile += gridDim.x * WAVES) {

        // A operand: lane l holds row (tile*16 + l&15); lanes 0-15 carry
        // K = {0..7, 16..23}, lanes 16-31 carry K = {8..15, 24..31}.
        const int    row   = (tile << 4) + lane15;
        const f32x4* aBase = (const f32x4*)(hidden + (size_t)row * H_DIM + (hi << 3));

        // B operand pointers: lane n holds expert column 16*j + (n&15);
        // lanes 0-15 carry K {0..7, 16..23}, lanes 16-31 carry K {8..15, 24..31} of each chunk.
        const unsigned short* b0p = &wlds[(lane15 +  0) * LDSS + (hi << 3)];
        const unsigned short* b1p = &wlds[(lane15 + 16) * LDSS + (hi << 3)];
        const unsigned short* b2p = &wlds[(lane15 + 32) * LDSS + (hi << 3)];
        const unsigned short* b3p = &wlds[(lane15 + 48) * LDSS + (hi << 3)];

        v8f acc0 = {}, acc1 = {}, acc2 = {}, acc3 = {};

        for (int kc = 0; kc < H_DIM; kc += 32) {
            const int kq = kc >> 2;       // f32x4 index

            // Streamed exactly once: non-temporal loads keep the L2 for weight.
            const f32x4 f0 = __builtin_nontemporal_load(aBase + kq + 0);
            const f32x4 f1 = __builtin_nontemporal_load(aBase + kq + 1);
            const f32x4 f2 = __builtin_nontemporal_load(aBase + kq + 4);
            const f32x4 f3 = __builtin_nontemporal_load(aBase + kq + 5);

            BF16x16 A;
            A.u[0] = pack_bf16(f0[0], f0[1]);
            A.u[1] = pack_bf16(f0[2], f0[3]);
            A.u[2] = pack_bf16(f1[0], f1[1]);
            A.u[3] = pack_bf16(f1[2], f1[3]);
            A.u[4] = pack_bf16(f2[0], f2[1]);
            A.u[5] = pack_bf16(f2[2], f2[3]);
            A.u[6] = pack_bf16(f3[0], f3[1]);
            A.u[7] = pack_bf16(f3[2], f3[3]);

            BF16x16 B0, B1, B2, B3;
            B0.q[0] = *(const uint4*)(b0p + kc); B0.q[1] = *(const uint4*)(b0p + kc + 16);
            B1.q[0] = *(const uint4*)(b1p + kc); B1.q[1] = *(const uint4*)(b1p + kc + 16);
            B2.q[0] = *(const uint4*)(b2p + kc); B2.q[1] = *(const uint4*)(b2p + kc + 16);
            B3.q[0] = *(const uint4*)(b3p + kc); B3.q[1] = *(const uint4*)(b3p + kc + 16);

            acc0 = __builtin_amdgcn_wmma_f32_16x16x32_bf16(false, A.v, false, B0.v,
                                                           (short)0, acc0, false, false);
            acc1 = __builtin_amdgcn_wmma_f32_16x16x32_bf16(false, A.v, false, B1.v,
                                                           (short)0, acc1, false, false);
            acc2 = __builtin_amdgcn_wmma_f32_16x16x32_bf16(false, A.v, false, B2.v,
                                                           (short)0, acc2, false, false);
            acc3 = __builtin_amdgcn_wmma_f32_16x16x32_bf16(false, A.v, false, B3.v,
                                                           (short)0, acc3, false, false);
        }

        // ---- Fused epilogue: branch-free per-row top-2 over 64 experts ----
        // C layout: VGPR slot m = row m (lanes 0-15) / row m+8 (lanes 16-31),
        // expert-within-tile = lane & 15.
        #pragma unroll
        for (int m = 0; m < 8; ++m) {
            // Per-lane top-2 of the 4 N-tiles (all selects, no branches).
            const unsigned long long k0 = mk_key(acc0[m], lane15 +  0);
            const unsigned long long k1 = mk_key(acc1[m], lane15 + 16);
            const unsigned long long k2 = mk_key(acc2[m], lane15 + 32);
            const unsigned long long k3 = mk_key(acc3[m], lane15 + 48);

            const unsigned long long a  = (k0 > k1) ? k0 : k1;
            const unsigned long long a2 = (k0 > k1) ? k1 : k0;
            const unsigned long long b  = (k2 > k3) ? k2 : k3;
            const unsigned long long b2 = (k2 > k3) ? k3 : k2;

            unsigned long long t = (a > b) ? a : b;                       // top
            unsigned long long s = (a > b) ? ((b > a2) ? b : a2)
                                           : ((a > b2) ? a : b2);         // second

            // Butterfly top-2 merge across the 16 lanes of each half-wave.
            #pragma unroll
            for (int mask = 1; mask < 16; mask <<= 1) {
                const unsigned long long ot =
                    (unsigned long long)__shfl_xor((long long)t, mask, 32);
                const unsigned long long os =
                    (unsigned long long)__shfl_xor((long long)s, mask, 32);
                const unsigned long long nt = (t > ot) ? t  : ot;
                const unsigned long long mn = (t > ot) ? ot : t;
                const unsigned long long c  = (t > ot) ? s  : os;
                s = (mn > c) ? mn : c;
                t = nt;
            }

            // One writer lane per row: lane m (rows 0-7), lane 16+m (rows 8-15).
            if (lane15 == m) {
                const int tok = (tile << 4) + m + (hi << 3);
                const float l0 = key_val(t);
                const float l1 = key_val(s);
                const float e1 = __expf(l1 - l0);        // <= 1, numerically safe
                const float p0 = 1.0f / (1.0f + e1);
                probs[2 * tok + 0]   = p0;
                probs[2 * tok + 1]   = 1.0f - p0;
                indices[2 * tok + 0] = key_idx(t);
                indices[2 * tok + 1] = key_idx(s);
            }
        }
    }
}

extern "C" void kernel_launch(void* const* d_in, const int* in_sizes, int n_in,
                              void* d_out, int out_size, void* d_ws, size_t ws_size,
                              hipStream_t stream) {
    const float* hidden = (const float*)d_in[0];   // [S*B, H] fp32
    const float* weight = (const float*)d_in[1];   // [E, H]  fp32

    const int T = in_sizes[0] / H_DIM;             // 16384 tokens

    // d_out = [probs: T*2 fp32][indices: T*2 int32] (tuple concatenated flat)
    float* probs   = (float*)d_out;
    int*   indices = (int*)d_out + (size_t)T * TOPK;

    // 64 blocks x 16 waves = 1024 wave-tiles = T/16 tiles exactly.
    topk_router_kernel<<<dim3(64), dim3(BLOCK), 0, stream>>>(
        hidden, weight, probs, indices, T);
}
